// EuclideanGraphDecoder_28114855919639
// MI455X (gfx1250) — matrix operand order repaired and checked by reading, not compiled, not run on hardware
//
#include <hip/hip_runtime.h>

typedef __attribute__((ext_vector_type(16))) __bf16 v16bf;
typedef __attribute__((ext_vector_type(8)))  float  v8f;
typedef __attribute__((ext_vector_type(4)))  float  fx4;
typedef __attribute__((ext_vector_type(4)))  unsigned int ux4;

#define B_   8
#define N_   2048
#define HID_ 128
#define OUT_ 64

// LDS B-tile: 128 columns x (32 K-elems + 8 pad) bf16, double buffered.
// 80B column stride keeps 16B alignment for b128 ops and staggers banks.
#define LDS_COL_STRIDE 40

// ---------------------------------------------------------------------------
// fp32 adjacency -> bf16 (8 elems/thread). Non-temporal loads: the 134MB fp32
// source is read exactly once; keep L2 free for the 67MB bf16 working set.
// ---------------------------------------------------------------------------
__global__ __launch_bounds__(256)
void cvt_bf16_kernel(const float* __restrict__ src, __bf16* __restrict__ dst,
                     long long n8) {
  long long i = (long long)blockIdx.x * blockDim.x + threadIdx.x;
  if (i >= n8) return;
  const fx4* s = (const fx4*)src + i * 2;
  fx4 a = __builtin_nontemporal_load(s);
  fx4 b = __builtin_nontemporal_load(s + 1);
  union { __bf16 h[8]; uint4 u; } o;
  o.h[0] = (__bf16)a.x; o.h[1] = (__bf16)a.y; o.h[2] = (__bf16)a.z; o.h[3] = (__bf16)a.w;
  o.h[4] = (__bf16)b.x; o.h[5] = (__bf16)b.y; o.h[6] = (__bf16)b.z; o.h[7] = (__bf16)b.w;
  *(uint4*)(dst + i * 8) = o.u;
}

// ---------------------------------------------------------------------------
// msg = h @ W + bias, stored TRANSPOSED as bf16: msgT[b][o][n]. <2% of FLOPs.
// ---------------------------------------------------------------------------
__global__ __launch_bounds__(256)
void linear_msgT_kernel(const float* __restrict__ hIn, const float* __restrict__ W,
                        const float* __restrict__ bias, __bf16* __restrict__ msgT,
                        int din) {
  int idx = blockIdx.x * 256 + threadIdx.x;      // B*HID*N threads, n fastest
  int n = idx & (N_ - 1);
  int o = (idx >> 11) & (HID_ - 1);
  int b = idx >> 18;
  const float* hr = hIn + ((size_t)b * N_ + n) * din;
  float s = bias[o];
  #pragma unroll 4
  for (int k = 0; k < din; ++k) s = fmaf(hr[k], W[k * HID_ + o], s);
  msgT[idx] = (__bf16)s;
}

// ---------------------------------------------------------------------------
// hOut = relu( adj(bf16, NxN) @ msg(bf16, Nx128) ), f32 accumulate.
//
// Block = 8 waves; each wave owns a 16x128 output strip (accum 64 VGPRs).
// ALL global memory traffic in the K-loop is issued via inline asm so LLVM's
// waitcnt tracker sees zero pending VMEM and never force-waits at asm
// boundaries. Waits are placed by hand:
//   * s_wait_asynccnt 0x2  -> current LDS B-buffer complete (async copy,
//     double buffered, next buffer's 2 ops stay in flight)
//   * s_wait_loadcnt 0x2 tied through a0/a1 -> A(t) fragment complete while
//     A(t+1) (issued this iteration) keeps flying: one full K-step of overlap.
// ---------------------------------------------------------------------------
__global__ __launch_bounds__(256)
void agg_wmma_kernel(const __bf16* __restrict__ adjB,   // [B][N][N]
                     const __bf16* __restrict__ msgT,   // [B][HID][N]
                     float* __restrict__ hOut) {        // [B][N][HID]
  __shared__ __bf16 ldsB[2][HID_ * LDS_COL_STRIDE];     // 2 x 10KB

  const int tid  = threadIdx.x;
  const int wave = tid >> 5;
  const int lane = tid & 31;
  const int tile = blockIdx.x * 8 + wave;               // 16-row tile id
  const int b    = tile >> 7;                           // 128 tiles per batch
  const int row0 = (tile & 127) << 4;

  const int half = lane >> 4;
  const int m    = lane & 15;

  const __bf16* aPtr    = adjB + ((size_t)b * N_ + row0 + m) * N_ + half * 8;
  const __bf16* msgBase = msgT + (size_t)b * HID_ * N_;

  // Async-copy mapping: each thread copies 16 contiguous bf16 of one column
  // (two b128 chunks; the inst offset applies to both LDS and global sides).
  const int ccol = tid >> 1;                 // 0..127
  const int q0   = (tid & 1) * 2;            // chunk pair 0/1 or 2/3
  const __bf16* gB = msgBase + (size_t)ccol * N_ + q0 * 8;
  const unsigned ldsW0 =
      (unsigned)(uintptr_t)&ldsB[0][ccol * LDS_COL_STRIDE + q0 * 8];
  const unsigned ldsW1 =
      (unsigned)(uintptr_t)&ldsB[1][ccol * LDS_COL_STRIDE + q0 * 8];

  v8f acc[8];
  #pragma unroll
  for (int j = 0; j < 8; ++j)
    #pragma unroll
    for (int v = 0; v < 8; ++v) acc[j][v] = 0.0f;

#define ISSUE_B2(LW, G)                                                        \
  do {                                                                         \
    asm volatile("global_load_async_to_lds_b128 %0, %1, off"                   \
                 :: "v"(LW), "v"(G));                                          \
    asm volatile("global_load_async_to_lds_b128 %0, %1, off offset:16"         \
                 :: "v"(LW), "v"(G));                                          \
  } while (0)

  // A-fragment loads via asm: invisible to the waitcnt tracker.
#define LOAD_A2(D0, D1, P)                                                     \
  asm volatile("global_load_b128 %0, %2, off\n\t"                              \
               "global_load_b128 %1, %2, off offset:32"                        \
               : "=&v"(D0), "=&v"(D1) : "v"(P))

#define COMPUTE_STEP(LBASE, AF)                                                \
  do {                                                                         \
    _Pragma("unroll")                                                          \
    for (int jj = 0; jj < 8; jj += 4) {                                        \
      v16bf bf[4];                                                             \
      _Pragma("unroll")                                                        \
      for (int j2 = 0; j2 < 4; ++j2) {                                         \
        const __bf16* lp = (LBASE) + ((jj + j2) * 16 + m) * LDS_COL_STRIDE;    \
        *(ux4*)&bf[j2]       = *(const ux4*)lp;                                \
        *((ux4*)&bf[j2] + 1) = *(const ux4*)(lp + 16);                         \
      }                                                                        \
      _Pragma("unroll")                                                        \
      for (int j2 = 0; j2 < 4; ++j2)                                           \
        acc[jj + j2] = __builtin_amdgcn_wmma_f32_16x16x32_bf16(                \
            false, (AF), false, bf[j2], (short)0, acc[jj + j2], false, false); \
    }                                                                          \
  } while (0)

  // Prologue: buffer 0 <- K[0..31]; A(0) in flight.
  ISSUE_B2(ldsW0, gB);
  ux4 a0, a1;
  LOAD_A2(a0, a1, aPtr);

  // Steady state: t = 0..62, branch-free body, unconditional rotation.
  for (int t = 0; t < 63; ++t) {
    const int kb = t * 32;

    // Issue next-buffer async copy + next A fragment; wait current B buffer.
    const __bf16* g = gB + kb + 32;
    const unsigned lw = (t & 1) ? ldsW0 : ldsW1;
    ISSUE_B2(lw, g);
    ux4 n0, n1;
    LOAD_A2(n0, n1, aPtr + kb + 32);
    asm volatile("s_wait_asynccnt 0x2");     // own 2 next-buf ops may fly on

    __syncthreads();                         // current buffer visible

    // A(t) done (4 pending, in-order: <=2 leaves only the A(t+1) pair).
    asm volatile("s_wait_loadcnt 0x2" : "+v"(a0), "+v"(a1));

    v16bf afrag;
    *(ux4*)&afrag       = a0;
    *((ux4*)&afrag + 1) = a1;

    const __bf16* lbase = &ldsB[t & 1][half * 8];
    COMPUTE_STEP(lbase, afrag);

    __syncthreads();                         // reads done before buffer refill
    a0 = n0;
    a1 = n1;
  }

  // Epilogue: last K-step (buffer 1), nothing left in flight afterwards.
  asm volatile("s_wait_asynccnt 0x0");
  __syncthreads();
  asm volatile("s_wait_loadcnt 0x0" : "+v"(a0), "+v"(a1));
  {
    v16bf afrag;
    *(ux4*)&afrag       = a0;
    *((ux4*)&afrag + 1) = a1;
    const __bf16* lbase = &ldsB[1][half * 8];
    COMPUTE_STEP(lbase, afrag);
  }
#undef COMPUTE_STEP
#undef LOAD_A2
#undef ISSUE_B2

  // Fused ReLU + store per C/D layout (VGPR v: M = half*8+v, N = m).
  float* oBase = hOut + ((size_t)b * N_ + row0) * HID_;
  #pragma unroll
  for (int j = 0; j < 8; ++j)
    #pragma unroll
    for (int v = 0; v < 8; ++v)
      oBase[(size_t)(half * 8 + v) * HID_ + j * 16 + m] = fmaxf(acc[j][v], 0.0f);
}

// ---------------------------------------------------------------------------
// out = (h @ Wout + bout) * node_mask, f32.
// ---------------------------------------------------------------------------
__global__ __launch_bounds__(256)
void proj_kernel(const float* __restrict__ h, const float* __restrict__ Wout,
                 const float* __restrict__ bout, const float* __restrict__ mask,
                 float* __restrict__ out) {
  int idx = blockIdx.x * 256 + threadIdx.x;      // B*N*OUT threads, o fastest
  int o = idx & (OUT_ - 1);
  int n = (idx >> 6) & (N_ - 1);
  int b = idx >> 17;
  const float* hr = h + ((size_t)b * N_ + n) * HID_;
  float s = bout[o];
  #pragma unroll 4
  for (int k = 0; k < HID_; ++k) s = fmaf(hr[k], Wout[k * OUT_ + o], s);
  out[idx] = s * mask[(size_t)b * N_ + n];
}

// ---------------------------------------------------------------------------
extern "C" void kernel_launch(void* const* d_in, const int* in_sizes, int n_in,
                              void* d_out, int out_size, void* d_ws, size_t ws_size,
                              hipStream_t stream) {
  (void)in_sizes; (void)n_in; (void)out_size; (void)ws_size;
  const float* latent = (const float*)d_in[0];
  const float* adj    = (const float*)d_in[1];
  const float* mask   = (const float*)d_in[2];
  const float* W0 = (const float*)d_in[3]; const float* b0 = (const float*)d_in[4];
  const float* W1 = (const float*)d_in[5]; const float* b1 = (const float*)d_in[6];
  const float* W2 = (const float*)d_in[7]; const float* b2 = (const float*)d_in[8];
  const float* Wo = (const float*)d_in[9]; const float* bo = (const float*)d_in[10];
  float* out = (float*)d_out;

  // Workspace: adjB bf16 [B][N][N] (64MiB) | msgT bf16 [B][HID][N] (4MiB)
  //            | h f32 [B][N][HID] (8MiB)
  size_t adjBytes = (size_t)B_ * N_ * N_ * 2;
  size_t msgBytes = (size_t)B_ * HID_ * N_ * 2;
  __bf16* adjB = (__bf16*)d_ws;
  __bf16* msgT = (__bf16*)((char*)d_ws + adjBytes);
  float*  h    = (float*)((char*)d_ws + adjBytes + msgBytes);

  long long adjN8 = (long long)B_ * N_ * N_ / 8;
  int cvtBlocks  = (int)((adjN8 + 255) / 256);
  int linBlocks  = (B_ * HID_ * N_) / 256;         // 8192
  int aggBlocks  = (B_ * (N_ / 16)) / 8;           // 128 blocks of 8 waves
  int projBlocks = (B_ * N_ * OUT_) / 256;         // 4096

  cvt_bf16_kernel<<<cvtBlocks, 256, 0, stream>>>(adj, adjB, adjN8);

  linear_msgT_kernel<<<linBlocks, 256, 0, stream>>>(latent, W0, b0, msgT, 64);
  agg_wmma_kernel<<<aggBlocks, 256, 0, stream>>>(adjB, msgT, h);

  linear_msgT_kernel<<<linBlocks, 256, 0, stream>>>(h, W1, b1, msgT, 128);
  agg_wmma_kernel<<<aggBlocks, 256, 0, stream>>>(adjB, msgT, h);

  linear_msgT_kernel<<<linBlocks, 256, 0, stream>>>(h, W2, b2, msgT, 128);
  agg_wmma_kernel<<<aggBlocks, 256, 0, stream>>>(adjB, msgT, h);

  proj_kernel<<<projBlocks, 256, 0, stream>>>(h, Wo, bo, mask, out);
}
